// SpatialBlock_48979807044334
// MI455X (gfx1250) — matrix operand order repaired and checked by reading, not compiled, not run on hardware
//
#include <hip/hip_runtime.h>
#include <hip/hip_bf16.h>

typedef __attribute__((ext_vector_type(16))) _Float16 v16h;
typedef __attribute__((ext_vector_type(8)))  _Float16 v8h;
typedef __attribute__((ext_vector_type(8)))  float    v8f;

#define TOK 100352ULL   // 32 * 56 * 56
#define CC  192
#define HIDC 768

__device__ __forceinline__ v16h cat8(v8h lo, v8h hi) {
    v16h r;
#pragma unroll
    for (int i = 0; i < 8; ++i) { r[i] = lo[i]; r[8 + i] = hi[i]; }
    return r;
}

__device__ __forceinline__ v8f wmma16(v16h a, v16h b, v8f c) {
    return __builtin_amdgcn_wmma_f32_16x16x32_f16(false, a, false, b, (short)0, c, false, false);
}

// ---- gfx1250 async global->LDS copy (bypasses VGPRs, tracked on ASYNCcnt) --
__device__ __forceinline__ void async_ld16(const _Float16* lds_dst, const _Float16* gsrc) {
    unsigned lds_off = (unsigned)(size_t)lds_dst;              // LDS aperture: addr[31:0]
    unsigned long long ga = (unsigned long long)(size_t)gsrc;
    asm volatile("global_load_async_to_lds_b128 %0, %1, off"
                 :: "v"(lds_off), "v"(ga) : "memory");
}
__device__ __forceinline__ void wait_async0() {
    asm volatile("s_wait_asynccnt 0x0" ::: "memory");
}

// ---------------- depthwise 3x3 CPE + residual (NHWC over [B, H*W, C]) -----
__global__ __launch_bounds__(256) void cpe_kernel(const float* __restrict__ x,
                                                  const float* __restrict__ w,
                                                  const float* __restrict__ bias,
                                                  float* __restrict__ out) {
    size_t idx = (size_t)blockIdx.x * blockDim.x + threadIdx.x;
    if (idx >= TOK * CC) return;
    int c = (int)(idx % CC);
    size_t r = idx / CC;
    int ww = (int)(r % 56); size_t r2 = r / 56;
    int hh = (int)(r2 % 56); int bb = (int)(r2 / 56);
    float acc = bias[c];
#pragma unroll
    for (int dy = 0; dy < 3; ++dy) {
        int y = hh + dy - 1;
        if ((unsigned)y >= 56u) continue;
#pragma unroll
        for (int dx = 0; dx < 3; ++dx) {
            int xx = ww + dx - 1;
            if ((unsigned)xx >= 56u) continue;
            acc += x[(((size_t)bb * 3136) + (size_t)y * 56 + xx) * CC + c] * w[c * 9 + dy * 3 + dx];
        }
    }
    out[idx] = x[idx] + acc;
}

// ---------------- LayerNorm -> f16 (one wave32 per token) ------------------
__global__ __launch_bounds__(256) void ln_kernel(const float* __restrict__ x,
                                                 const float* __restrict__ g,
                                                 const float* __restrict__ b,
                                                 _Float16* __restrict__ out) {
    int lane = threadIdx.x & 31;
    size_t t = (size_t)blockIdx.x * 8 + (threadIdx.x >> 5);
    const float* xr = x + t * CC;
    float v[6];
    float s = 0.f;
#pragma unroll
    for (int j = 0; j < 6; ++j) { v[j] = xr[lane + 32 * j]; s += v[j]; }
#pragma unroll
    for (int o = 16; o > 0; o >>= 1) s += __shfl_xor(s, o, 32);
    float mean = s * (1.f / 192.f);
    float vs = 0.f;
#pragma unroll
    for (int j = 0; j < 6; ++j) { float d = v[j] - mean; vs += d * d; }
#pragma unroll
    for (int o = 16; o > 0; o >>= 1) vs += __shfl_xor(vs, o, 32);
    float rs = rsqrtf(vs * (1.f / 192.f) + 1e-5f);
#pragma unroll
    for (int j = 0; j < 6; ++j) {
        int c = lane + 32 * j;
        out[t * CC + c] = (_Float16)((v[j] - mean) * rs * g[c] + b[c]);
    }
}

// ---------------- weight transpose/convert: W[K][N] f32 -> Bt[N][K] f16 ----
__global__ __launch_bounds__(256) void wt_kernel(const float* __restrict__ W,
                                                 _Float16* __restrict__ Bt,
                                                 int K, int N) {
    int idx = blockIdx.x * blockDim.x + threadIdx.x;
    if (idx >= K * N) return;
    int k = idx / N, n = idx % N;
    Bt[(size_t)n * K + k] = (_Float16)W[idx];
}

// ---------------- WMMA GEMM: 128x64 tile / block, double-buffered LDS ------
// 8 waves: wave w owns a 16-row M strip and all four 16-col N subtiles.
// Next K-slice streams into the ping-pong buffer via async global->LDS copies
// (no VGPR round-trip); the WMMAs on the current buffer overlap the stream.
// EPI: 0 plain f16 out, 1 GELU f16 out, 2 bias+residual f32 out,
//      3 qkv-permuted f16 out (col j=d*18+kk*6+h -> kk*192+h*32+d)
template <int KDIM, int EPI>
__global__ __launch_bounds__(256) void gemm_kernel(const _Float16* __restrict__ A,
                                                   const _Float16* __restrict__ Bt,
                                                   const float* __restrict__ bias,
                                                   const float* __restrict__ resid,
                                                   _Float16* __restrict__ outh,
                                                   float* __restrict__ outf,
                                                   int N) {
    __shared__ _Float16 Asl[2][128][32];
    __shared__ _Float16 Bsl[2][64][32];
    const int tid  = threadIdx.x;
    const int n0   = blockIdx.x * 64;
    const size_t m0 = (size_t)blockIdx.y * 128;
    // staging: A 16 halves/thread (128x32), B 8 halves/thread (64x32)
    const int am = tid >> 1;          // 0..127
    const int ak = (tid & 1) * 16;    // 0 / 16
    const int bn = tid >> 2;          // 0..63
    const int bk = (tid & 3) * 8;     // 0,8,16,24
    const int lane = tid & 31, w = tid >> 5;
    const int fr = lane & 15, fc = lane >> 4;

    const _Float16* Aptr = A + (m0 + am) * KDIM + ak;
    const _Float16* Bptr = Bt + (size_t)(n0 + bn) * KDIM + bk;

    v8f acc[4];
#pragma unroll
    for (int nt = 0; nt < 4; ++nt) acc[nt] = (v8f){};

    // prime buffer 0 (async copies straight into LDS)
    async_ld16(&Asl[0][am][ak],     Aptr);
    async_ld16(&Asl[0][am][ak + 8], Aptr + 8);
    async_ld16(&Bsl[0][bn][bk],     Bptr);
    wait_async0();
    __syncthreads();

    const int nIter = KDIM / 32;
#pragma unroll 2
    for (int it = 0; it < nIter; ++it) {
        const int cur = it & 1;
        const bool more = (it + 1) < nIter;
        if (more) {  // stream next K-slice into the other buffer while we compute
            const int nxt = cur ^ 1;
            async_ld16(&Asl[nxt][am][ak],     Aptr + (it + 1) * 32);
            async_ld16(&Asl[nxt][am][ak + 8], Aptr + (it + 1) * 32 + 8);
            async_ld16(&Bsl[nxt][bn][bk],     Bptr + (it + 1) * 32);
        }
        v16h af = cat8(*(const v8h*)&Asl[cur][w * 16 + fr][8 * fc],
                       *(const v8h*)&Asl[cur][w * 16 + fr][16 + 8 * fc]);
#pragma unroll
        for (int nt = 0; nt < 4; ++nt) {
            v16h bf = cat8(*(const v8h*)&Bsl[cur][nt * 16 + fr][8 * fc],
                           *(const v8h*)&Bsl[cur][nt * 16 + fr][16 + 8 * fc]);
            acc[nt] = wmma16(af, bf, acc[nt]);
        }
        if (more) {
            wait_async0();     // this wave's async copies into nxt are complete
            __syncthreads();   // publishes nxt; all reads of cur are done
        }
    }

#pragma unroll
    for (int r = 0; r < 8; ++r) {
        size_t mm = m0 + w * 16 + 8 * fc + r;
#pragma unroll
        for (int nt = 0; nt < 4; ++nt) {
            int nn = n0 + nt * 16 + fr;
            float val = acc[nt][r] + bias[nn];
            if constexpr (EPI == 0) {
                outh[mm * (size_t)N + nn] = (_Float16)val;
            } else if constexpr (EPI == 1) {
                float gv = 0.5f * val * (1.f + erff(val * 0.70710678f));
                outh[mm * (size_t)N + nn] = (_Float16)gv;
            } else if constexpr (EPI == 2) {
                outf[mm * (size_t)N + nn] = val + resid[mm * (size_t)N + nn];
            } else {
                int d = nn / 18, rr = nn - 18 * d;
                int kk = rr / 6, hh = rr - 6 * kk;
                outh[mm * 576 + kk * 192 + hh * 32 + d] = (_Float16)val;
            }
        }
    }
}

// ---------------- window attention: one block = one (window, head) ---------
// qkv: [T][3][6][32] f16 (permuted by the QKV epilogue); attn_out: [T][192] f16
__global__ __launch_bounds__(64) void attn_kernel(const _Float16* __restrict__ qkv,
                                                  _Float16* __restrict__ attn_out) {
    __shared__ _Float16 Qs[64][32];
    __shared__ _Float16 Ks[64][32];
    __shared__ _Float16 Vt[32][64];
    __shared__ _Float16 Ps[64][64];

    int bx = blockIdx.x;
    int h  = bx % 6;
    int wi = bx / 6;
    int bimg = wi >> 6;
    int ih = (wi >> 3) & 7;
    int iw = wi & 7;
    int tid = threadIdx.x;

    // stage Q, K, V (dilated-window mapping: row = wh*8+ih, col = ww*8+iw)
    {
        int n = tid;
        if (n < 49) {
            int wh = n / 7, ww = n % 7;
            size_t t = (size_t)bimg * 3136 + (size_t)(wh * 8 + ih) * 56 + (ww * 8 + iw);
            const _Float16* q = qkv + t * 576 + h * 32;
            const _Float16* k = q + 192;
            const _Float16* v = q + 384;
#pragma unroll
            for (int j = 0; j < 32; j += 8) {
                *(v8h*)&Qs[n][j] = *(const v8h*)(q + j);
                *(v8h*)&Ks[n][j] = *(const v8h*)(k + j);
            }
#pragma unroll
            for (int d = 0; d < 32; ++d) Vt[d][n] = v[d];
        } else {
#pragma unroll
            for (int j = 0; j < 32; ++j) { Qs[n][j] = (_Float16)0.f; Ks[n][j] = (_Float16)0.f; }
#pragma unroll
            for (int d = 0; d < 32; ++d) Vt[d][n] = (_Float16)0.f;
        }
    }
    __syncthreads();

    int lane = tid & 31, wv = tid >> 5;
    int fr = lane & 15, fc = lane >> 4;

    // scores S = Q K^T (64x64 padded), wave wv owns m-tiles {2wv, 2wv+1}
    v8f s[2][4];
#pragma unroll
    for (int i = 0; i < 2; ++i)
#pragma unroll
        for (int nt = 0; nt < 4; ++nt) s[i][nt] = (v8f){};

    v16h qf[2];
#pragma unroll
    for (int i = 0; i < 2; ++i) {
        int m = (2 * wv + i) * 16 + fr;
        qf[i] = cat8(*(const v8h*)&Qs[m][8 * fc], *(const v8h*)&Qs[m][16 + 8 * fc]);
    }
#pragma unroll
    for (int nt = 0; nt < 4; ++nt) {
        int n = nt * 16 + fr;
        v16h kf = cat8(*(const v8h*)&Ks[n][8 * fc], *(const v8h*)&Ks[n][16 + 8 * fc]);
#pragma unroll
        for (int i = 0; i < 2; ++i) s[i][nt] = wmma16(qf[i], kf, s[i][nt]);
    }

    // row softmax with pad masking (cols >= 49 -> -inf); write P (normalized) to LDS
    const float scale = 0.17677669529663687f;  // 32^-0.5
    bool colok[4];
#pragma unroll
    for (int nt = 0; nt < 4; ++nt) colok[nt] = (nt * 16 + fr) < 49;

#pragma unroll
    for (int i = 0; i < 2; ++i) {
#pragma unroll
        for (int r = 0; r < 8; ++r) {
            float sv[4];
            float mx = -1e30f;
#pragma unroll
            for (int nt = 0; nt < 4; ++nt) {
                sv[nt] = colok[nt] ? s[i][nt][r] * scale : -1e30f;
                mx = fmaxf(mx, sv[nt]);
            }
#pragma unroll
            for (int o = 8; o > 0; o >>= 1) mx = fmaxf(mx, __shfl_xor(mx, o, 32));
            float sum = 0.f;
#pragma unroll
            for (int nt = 0; nt < 4; ++nt) { sv[nt] = __expf(sv[nt] - mx); sum += sv[nt]; }
#pragma unroll
            for (int o = 8; o > 0; o >>= 1) sum += __shfl_xor(sum, o, 32);
            float inv = 1.f / sum;
            int m = (2 * wv + i) * 16 + 8 * fc + r;
#pragma unroll
            for (int nt = 0; nt < 4; ++nt) Ps[m][nt * 16 + fr] = (_Float16)(sv[nt] * inv);
        }
    }
    __syncthreads();

    // O = P V  (64x32), K dim 64 -> two WMMAs per output tile
    v8f o[2][2];
#pragma unroll
    for (int i = 0; i < 2; ++i)
#pragma unroll
        for (int nt = 0; nt < 2; ++nt) o[i][nt] = (v8f){};

#pragma unroll
    for (int kk = 0; kk < 2; ++kk) {
#pragma unroll
        for (int i = 0; i < 2; ++i) {
            int m = (2 * wv + i) * 16 + fr;
            v16h pf = cat8(*(const v8h*)&Ps[m][kk * 32 + 8 * fc],
                           *(const v8h*)&Ps[m][kk * 32 + 16 + 8 * fc]);
#pragma unroll
            for (int nt = 0; nt < 2; ++nt) {
                int d = nt * 16 + fr;
                v16h vf = cat8(*(const v8h*)&Vt[d][kk * 32 + 8 * fc],
                               *(const v8h*)&Vt[d][kk * 32 + 16 + 8 * fc]);
                o[i][nt] = wmma16(pf, vf, o[i][nt]);
            }
        }
    }

    // scatter back to token order, concatenated heads
#pragma unroll
    for (int i = 0; i < 2; ++i) {
#pragma unroll
        for (int r = 0; r < 8; ++r) {
            int m = (2 * wv + i) * 16 + 8 * fc + r;
            if (m < 49) {
                int wh = m / 7, ww = m % 7;
                size_t t = (size_t)bimg * 3136 + (size_t)(wh * 8 + ih) * 56 + (ww * 8 + iw);
#pragma unroll
                for (int nt = 0; nt < 2; ++nt) {
                    int d = nt * 16 + fr;
                    attn_out[t * CC + h * 32 + d] = (_Float16)o[i][nt][r];
                }
            }
        }
    }
}

extern "C" void kernel_launch(void* const* d_in, const int* in_sizes, int n_in,
                              void* d_out, int out_size, void* d_ws, size_t ws_size,
                              hipStream_t stream) {
    const float* x      = (const float*)d_in[0];
    const float* cpe0_w = (const float*)d_in[3];
    const float* cpe0_b = (const float*)d_in[4];
    const float* cpe1_w = (const float*)d_in[5];
    const float* cpe1_b = (const float*)d_in[6];
    const float* n1_g   = (const float*)d_in[7];
    const float* n1_b   = (const float*)d_in[8];
    const float* qkv_w  = (const float*)d_in[9];
    const float* qkv_b  = (const float*)d_in[10];
    const float* proj_w = (const float*)d_in[11];
    const float* proj_b = (const float*)d_in[12];
    const float* n2_g   = (const float*)d_in[13];
    const float* n2_b   = (const float*)d_in[14];
    const float* fc1_w  = (const float*)d_in[15];
    const float* fc1_b  = (const float*)d_in[16];
    const float* fc2_w  = (const float*)d_in[17];
    const float* fc2_b  = (const float*)d_in[18];
    float* out = (float*)d_out;

    const size_t F32SZ = TOK * CC * 4ULL;      // 77,070,336
    const size_t F16SZ = TOK * CC * 2ULL;      // 38,535,168
    const size_t QKVSZ = TOK * 3ULL * CC * 2;  // 115,605,504

    char* ws = (char*)d_ws;
    float*     shortcut = (float*)ws;
    float*     ybuf     = (float*)(ws + F32SZ);
    _Float16*  ln16     = (_Float16*)(ws + 2 * F32SZ);
    _Float16*  qkvb     = (_Float16*)(ws + 2 * F32SZ + F16SZ);
    _Float16*  attn_o   = (_Float16*)((char*)qkvb + QKVSZ);
    _Float16*  hbuf     = qkvb;  // FC1 output reuses dead qkv+attn space (exact fit)
    _Float16*  wq = (_Float16*)((char*)attn_o + F16SZ);
    _Float16*  wp = wq + 576 * 192;
    _Float16*  w1 = wp + 192 * 192;
    _Float16*  w2 = w1 + 192 * 768;

    // weight transpose/convert to f16 [N][K]
    wt_kernel<<<(192 * 576 + 255) / 256, 256, 0, stream>>>(qkv_w, wq, 192, 576);
    wt_kernel<<<(192 * 192 + 255) / 256, 256, 0, stream>>>(proj_w, wp, 192, 192);
    wt_kernel<<<(192 * 768 + 255) / 256, 256, 0, stream>>>(fc1_w, w1, 192, 768);
    wt_kernel<<<(768 * 192 + 255) / 256, 256, 0, stream>>>(fc2_w, w2, 768, 192);

    const int elemBlocks = (int)((TOK * CC + 255) / 256);
    const int mTiles = (int)(TOK / 128);  // 784
    const int lnBlocks = (int)(TOK / 8);  // 12544

    // shortcut = x + CPE0(x)
    cpe_kernel<<<elemBlocks, 256, 0, stream>>>(x, cpe0_w, cpe0_b, shortcut);
    // ln16 = LN1(shortcut) in f16
    ln_kernel<<<lnBlocks, 256, 0, stream>>>(shortcut, n1_g, n1_b, ln16);
    // qkvb = ln16 @ qkv_w + qkv_b (permuted to [3][NH][hd])
    gemm_kernel<192, 3><<<dim3(9, mTiles), 256, 0, stream>>>(ln16, wq, qkv_b, nullptr, qkvb, nullptr, 576);
    // window attention
    attn_kernel<<<32 * 64 * 6, 64, 0, stream>>>(qkvb, attn_o);
    // ybuf = shortcut + attn_o @ proj_w + proj_b
    gemm_kernel<192, 2><<<dim3(3, mTiles), 256, 0, stream>>>(attn_o, wp, proj_b, shortcut, nullptr, ybuf, 192);
    // out(=y2) = ybuf + CPE1(ybuf)
    cpe_kernel<<<elemBlocks, 256, 0, stream>>>(ybuf, cpe1_w, cpe1_b, out);
    // ln16 = LN2(y2)
    ln_kernel<<<lnBlocks, 256, 0, stream>>>(out, n2_g, n2_b, ln16);
    // hbuf = gelu(ln16 @ fc1_w + fc1_b)
    gemm_kernel<192, 1><<<dim3(12, mTiles), 256, 0, stream>>>(ln16, w1, fc1_b, nullptr, hbuf, nullptr, 768);
    // out = y2 + hbuf @ fc2_w + fc2_b
    gemm_kernel<768, 2><<<dim3(3, mTiles), 256, 0, stream>>>(hbuf, w2, fc2_b, out, nullptr, out, 192);
}
